// GaussianRasterizer_90821378441733
// MI455X (gfx1250) — matrix operand order repaired and checked by reading, not compiled, not run on hardware
//
#include <hip/hip_runtime.h>

#define HH 128
#define WW 256
#define HWP (HH*WW)          // 32768 pixels
#define NG 1024
#define NF 16
#define NCH 20               // 16 feat + rgb + depth
#define NCHUNK (NG/32)
#define FXc 128.0f           // W/(2*tanfovx)
#define FYc 128.0f           // H/(2*tanfovy)
#define ALPHA_T (1.0f/255.0f)
#define T_T 1.0e-4f

typedef __attribute__((ext_vector_type(16))) _Float16 v16h;
typedef __attribute__((ext_vector_type(8)))  float    v8f;
typedef unsigned int u32v4 __attribute__((ext_vector_type(4)));
typedef int          i32v8 __attribute__((ext_vector_type(8)));
typedef int          i32v4 __attribute__((ext_vector_type(4)));

// Workspace layout (float offsets):
//  [0*NG..9*NG)  unsorted per-gaussian: key,mpx,mpy,ca,cb,cc,op,rad,tz
//  PARAM_OFF     sorted params as dense [7][NG] f32 tensor (rows: mpx,mpy,ca,cb,cc,op,rad)
//  CHAN_OFF      sorted channels as dense [NG][20] f16 tensor (feat0..15, r,g,b, tz)
#define WS_KEY 0
#define WS_MPX 1
#define WS_MPY 2
#define WS_CA  3
#define WS_CB  4
#define WS_CC  5
#define WS_OP  6
#define WS_RAD 7
#define WS_TZ  8
#define PARAM_OFF 10240
#define CHAN_OFF  17408      // float offset; f16 array of NG*20 elements lives here

// ---- TDM issue helper: 2D tile load (CDNA5 ISA).
// This toolchain exposes the 6-arg builtin:
//   (uint32x4 g0, int32x8 g1, int32x4 g2, int32x4 g3, int32x8 ext, i32 cpol)
__device__ __forceinline__ void tdm_load_2d(unsigned lds_off, unsigned long long gaddr,
                                            unsigned data_size, unsigned td0, unsigned td1,
                                            unsigned tl0, unsigned tl1,
                                            unsigned long long stride0)
{
  u32v4 g0;
  g0[0] = 1u;                                             // count=1, user mode
  g0[1] = lds_off;                                        // lds_addr
  g0[2] = (unsigned)(gaddr & 0xFFFFFFFFull);              // global_addr[31:0]
  g0[3] = (unsigned)((gaddr >> 32) & 0x01FFFFFFull)       // global_addr[56:32]
        | (2u << 30);                                     // type = 2 ("image")
  i32v8 g1;
  g1[0] = (int)(data_size << 16);                         // workgroup_mask=0, data_size
  g1[1] = (int)((td0 & 0xFFFFu) << 16);                   // tensor_dim0[15:0] @ bit48
  g1[2] = (int)((td0 >> 16) | ((td1 & 0xFFFFu) << 16));   // td0[31:16] | td1[15:0]
  g1[3] = (int)((td1 >> 16) | (tl0 << 16));               // td1[31:16] | tile_dim0
  g1[4] = (int)(tl1 & 0xFFFFu);                           // tile_dim1 (0 = 1D), tile_dim2=0
  g1[5] = (int)(stride0 & 0xFFFFFFFFull);                 // tensor_dim0_stride[31:0]
  g1[6] = (int)((stride0 >> 32) & 0xFFFFull);             // stride0[47:32] | stride1[15:0]=0
  g1[7] = 0;                                              // stride1[47:16]=0
  i32v4 z4 = {};
  i32v8 z8 = {};
  __builtin_amdgcn_tensor_load_to_lds(g0, g1, z4, z4, z8, 0);
}

// ---------------------------------------------------------------- preprocess
__global__ __launch_bounds__(256) void gs_preprocess(
    const float* __restrict__ means, const float* __restrict__ opac,
    const float* __restrict__ scales, const float* __restrict__ rots,
    const float* __restrict__ vm, const unsigned char* __restrict__ mask,
    float* __restrict__ ws, float* __restrict__ outRadii)
{
  int i = blockIdx.x * blockDim.x + threadIdx.x;
  if (i >= NG) return;

  float qw = rots[4*i+0], qx = rots[4*i+1], qy = rots[4*i+2], qz = rots[4*i+3];
  float qn = rsqrtf(qw*qw + qx*qx + qy*qy + qz*qz);
  qw *= qn; qx *= qn; qy *= qn; qz *= qn;
  float r00 = 1.f-2.f*(qy*qy+qz*qz), r01 = 2.f*(qx*qy-qw*qz), r02 = 2.f*(qx*qz+qw*qy);
  float r10 = 2.f*(qx*qy+qw*qz), r11 = 1.f-2.f*(qx*qx+qz*qz), r12 = 2.f*(qy*qz-qw*qx);
  float r20 = 2.f*(qx*qz-qw*qy), r21 = 2.f*(qy*qz+qw*qx), r22 = 1.f-2.f*(qx*qx+qy*qy);

  float s0 = scales[3*i+0], s1 = scales[3*i+1], s2 = scales[3*i+2];
  float m00 = r00*s0, m01 = r01*s1, m02 = r02*s2;
  float m10 = r10*s0, m11 = r11*s1, m12 = r12*s2;
  float m20 = r20*s0, m21 = r21*s1, m22 = r22*s2;
  float c00 = m00*m00 + m01*m01 + m02*m02;
  float c01 = m00*m10 + m01*m11 + m02*m12;
  float c02 = m00*m20 + m01*m21 + m02*m22;
  float c11 = m10*m10 + m11*m11 + m12*m12;
  float c12 = m10*m20 + m11*m21 + m12*m22;
  float c22 = m20*m20 + m21*m21 + m22*m22;

  float p0 = means[3*i+0], p1 = means[3*i+1], p2 = means[3*i+2];
  float tx = p0*vm[0] + p1*vm[4] + p2*vm[8]  + vm[12];
  float ty = p0*vm[1] + p1*vm[5] + p2*vm[9]  + vm[13];
  float tz = p0*vm[2] + p1*vm[6] + p2*vm[10] + vm[14];

  float limx = 1.3f, limy = 0.65f;
  float txz = fminf(fmaxf(tx/tz, -limx), limx) * tz;
  float tyz = fminf(fmaxf(ty/tz, -limy), limy) * tz;
  float J00 = FXc/tz, J02 = -FXc*txz/(tz*tz);
  float J11 = FYc/tz, J12 = -FYc*tyz/(tz*tz);
  float a0 = J00*vm[0] + J02*vm[2];
  float a1 = J00*vm[4] + J02*vm[6];
  float a2 = J00*vm[8] + J02*vm[10];
  float b0 = J11*vm[1] + J12*vm[2];
  float b1 = J11*vm[5] + J12*vm[6];
  float b2 = J11*vm[9] + J12*vm[10];
  float u0 = a0*c00 + a1*c01 + a2*c02;
  float u1 = a0*c01 + a1*c11 + a2*c12;
  float u2 = a0*c02 + a1*c12 + a2*c22;
  float w0 = b0*c00 + b1*c01 + b2*c02;
  float w1 = b0*c01 + b1*c11 + b2*c12;
  float w2 = b0*c02 + b1*c12 + b2*c22;
  float av = u0*a0 + u1*a1 + u2*a2 + 0.3f;
  float bv = u0*b0 + u1*b1 + u2*b2;
  float cv = w0*b0 + w1*b1 + w2*b2 + 0.3f;

  float det = av*cv - bv*bv;
  bool vis = (tz > 0.2f) && (det > 0.0f) && (mask[i] != 0);
  float inv_det = (det != 0.0f) ? 1.0f/det : 0.0f;
  float ca = cv*inv_det, cb = -bv*inv_det, cc = av*inv_det;
  float mid = 0.5f*(av+cv);
  float lam = mid + sqrtf(fmaxf(mid*mid - det, 0.1f));
  float radius = vis ? ceilf(3.0f*sqrtf(lam)) : 0.0f;
  float mpx = FXc*tx/tz + (WW-1)*0.5f;
  float mpy = FYc*ty/tz + (HH-1)*0.5f;

  ws[WS_KEY*NG + i] = vis ? tz : 1.0e30f;
  ws[WS_MPX*NG + i] = mpx;
  ws[WS_MPY*NG + i] = mpy;
  ws[WS_CA *NG + i] = ca;
  ws[WS_CB *NG + i] = cb;
  ws[WS_CC *NG + i] = cc;
  ws[WS_OP *NG + i] = opac[i];
  ws[WS_RAD*NG + i] = vis ? radius : -1.0f;   // -1 sentinel: in_rect always false
  ws[WS_TZ *NG + i] = tz;
  outRadii[i] = radius;
}

// ------------------------------------------- bitonic sort + one-time gather
__global__ __launch_bounds__(1024) void gs_sort_gather(
    float* __restrict__ ws, const float* __restrict__ feats,
    const float* __restrict__ cols)
{
  __shared__ float sk[NG];
  __shared__ int   sv[NG];
  int t = threadIdx.x;
  sk[t] = ws[WS_KEY*NG + t];
  sv[t] = t;
  __syncthreads();
  for (int k = 2; k <= NG; k <<= 1) {
    for (int j = k >> 1; j > 0; j >>= 1) {
      int ixj = t ^ j;
      if (ixj > t) {
        bool up = ((t & k) == 0);
        float ka = sk[t], kb2 = sk[ixj];
        if (up ? (ka > kb2) : (ka < kb2)) {
          sk[t] = kb2; sk[ixj] = ka;
          int va = sv[t]; sv[t] = sv[ixj]; sv[ixj] = va;
        }
      }
      __syncthreads();
    }
  }
  // Gather into dense TDM-friendly tensors (done once, not per raster block).
  int s = sv[t];
  float* P = ws + PARAM_OFF;                 // [7][NG] f32
  P[0*NG + t] = ws[WS_MPX*NG + s];
  P[1*NG + t] = ws[WS_MPY*NG + s];
  P[2*NG + t] = ws[WS_CA *NG + s];
  P[3*NG + t] = ws[WS_CB *NG + s];
  P[4*NG + t] = ws[WS_CC *NG + s];
  P[5*NG + t] = ws[WS_OP *NG + s];
  P[6*NG + t] = ws[WS_RAD*NG + s];
  _Float16* C = (_Float16*)(ws + CHAN_OFF);  // [NG][20] f16
  #pragma unroll
  for (int m = 0; m < NF; ++m)
    C[t*NCH + m] = (_Float16)feats[s*NF + m];
  C[t*NCH + 16] = (_Float16)cols[s*3+0];
  C[t*NCH + 17] = (_Float16)cols[s*3+1];
  C[t*NCH + 18] = (_Float16)cols[s*3+2];
  C[t*NCH + 19] = (_Float16)ws[WS_TZ*NG + s];
}

// ---------------------------------------------------------------- rasterize
__global__ __launch_bounds__(256) void gs_raster(
    const float* __restrict__ ws, const float* __restrict__ bg,
    float* __restrict__ out)
{
  // Double-buffered LDS tiles filled by the Tensor Data Mover.
  __shared__ alignas(16) float    s_params[2][7][32];   // mpx,mpy,ca,cb,cc,op,rad
  __shared__ alignas(16) _Float16 s_chan[2][32][NCH];   // feat0..15, r,g,b, tz
  __shared__ alignas(64) _Float16 wbuf[8][32][32];      // [wave][pixel-in-wave][k]
  __shared__ float s_tl[256];

  const int tid  = threadIdx.x;
  const int wave = tid >> 5;
  const int lane = tid & 31;
  const int p    = blockIdx.x * 256 + tid;
  const float px = (float)(p & (WW - 1));
  const float py = (float)(p >> 8);

  const unsigned long long param_base =
      (unsigned long long)(size_t)(ws + PARAM_OFF);
  const unsigned long long chan_base =
      (unsigned long long)(size_t)(ws + CHAN_OFF);

  float Tb = 1.0f, Tl = 1.0f;
  int contrib = 0;
  v8f accF0 = {}, accF1 = {}, accC0 = {}, accC1 = {};

  // Prologue: wave 0 launches the TDM for chunk 0.
  if (wave == 0) {
    unsigned lp = (unsigned)(size_t)&s_params[0][0][0];   // LDS aperture: addr[31:0]
    unsigned lc = (unsigned)(size_t)&s_chan[0][0][0];
    tdm_load_2d(lp, param_base, 2u, NG, 7u, 32u, 7u, NG);                 // 7x32 f32, stride NG
    tdm_load_2d(lc, chan_base, 1u, NG*NCH, 1u, 32u*NCH, 0u, NG*NCH);      // 1280B contiguous
  }

  for (int chunk = 0; chunk < NCHUNK; ++chunk) {
    const int b = chunk & 1;
    if (wave == 0) __builtin_amdgcn_s_wait_tensorcnt(0);  // chunk's tiles landed in LDS
    __syncthreads();                                      // publish to all waves
    if (wave == 0 && chunk + 1 < NCHUNK) {                // prefetch next chunk via TDM
      const int nb = (chunk + 1) & 1;
      unsigned lp = (unsigned)(size_t)&s_params[nb][0][0];
      unsigned lc = (unsigned)(size_t)&s_chan[nb][0][0];
      tdm_load_2d(lp, param_base + (unsigned long long)(chunk+1)*128ull,
                  2u, NG, 7u, 32u, 7u, NG);
      tdm_load_2d(lc, chan_base + (unsigned long long)(chunk+1)*1280ull,
                  1u, NG*NCH, 1u, 32u*NCH, 0u, NG*NCH);
    }

    // Sequential transmittance recurrence (per-lane pixel) over 32 sorted gaussians.
    #pragma unroll 4
    for (int k = 0; k < 32; ++k) {
      float dx = px - s_params[b][0][k];
      float dy = py - s_params[b][1][k];
      float power = -(0.5f*s_params[b][2][k]*dx*dx + 0.5f*s_params[b][4][k]*dy*dy
                      + s_params[b][3][k]*dx*dy);
      float alpha = fminf(0.99f, s_params[b][5][k] * __expf(power));
      float r = s_params[b][6][k];
      bool keep = (fabsf(dx) <= r) && (fabsf(dy) <= r) &&
                  (power <= 0.0f) && (alpha >= ALPHA_T);
      alpha = keep ? alpha : 0.0f;
      float om = 1.0f - alpha;
      bool live = (Tb * om >= T_T);
      float wgt = live ? alpha * Tb : 0.0f;
      if (wgt > 0.0f) contrib++;
      Tl = live ? Tl * om : Tl;
      Tb *= om;
      wbuf[wave][lane][k] = (_Float16)wgt;
    }
    __syncthreads();

    // WMMA operands (16-bit A 16x32 / B 32x16, contiguous k = 16*(lane/16) + e in both A and B).
    const int m  = lane & 15;
    const int kb = (lane < 16) ? 0 : 8;
    const int kh = (lane < 16) ? 0 : 16;
    v16h aF, aC;
    #pragma unroll
    for (int e = 0; e < 16; ++e) {
      int ka = kh + e;                                   // K index for element e
      aF[e] = s_chan[b][ka][m];
      aC[e] = (m < 4) ? s_chan[b][ka][16 + m] : (_Float16)0.0f;
    }
    v16h b0 = *(const v16h*)(&wbuf[wave][m][kh]);        // pixels 0..15 of wave
    v16h b1 = *(const v16h*)(&wbuf[wave][16 + m][kh]);   // pixels 16..31 of wave

    accF0 = __builtin_amdgcn_wmma_f32_16x16x32_f16(false, aF, false, b0, (short)0, accF0, false, false);
    accF1 = __builtin_amdgcn_wmma_f32_16x16x32_f16(false, aF, false, b1, (short)0, accF1, false, false);
    accC0 = __builtin_amdgcn_wmma_f32_16x16x32_f16(false, aC, false, b0, (short)0, accC0, false, false);
    accC1 = __builtin_amdgcn_wmma_f32_16x16x32_f16(false, aC, false, b1, (short)0, accC1, false, false);
  }

  // Per-pixel scalar outputs.
  s_tl[tid] = Tl;
  out[p]          = (float)contrib;          // contrib
  out[21*HWP + p] = 1.0f - Tl;               // alpha map
  __syncthreads();

  const int wbase = blockIdx.x * 256 + wave * 32;
  const int m   = lane & 15;
  const int cb0 = (lane < 16) ? 0 : 8;
  const int pc0 = wbase + m;
  const int pc1 = wbase + 16 + m;
  #pragma unroll
  for (int r = 0; r < 8; ++r) {
    out[4*HWP + (cb0 + r)*HWP + pc0] = accF0[r];
    out[4*HWP + (cb0 + r)*HWP + pc1] = accF1[r];
  }
  if (lane < 16) {
    float tl0 = s_tl[wave*32 + lane];
    float tl1 = s_tl[wave*32 + 16 + lane];
    #pragma unroll
    for (int c = 0; c < 3; ++c) {
      out[HWP + c*HWP + pc0] = accC0[c] + tl0 * bg[c];
      out[HWP + c*HWP + pc1] = accC1[c] + tl1 * bg[c];
    }
    out[20*HWP + pc0] = accC0[3];            // depth
    out[20*HWP + pc1] = accC1[3];
  }
}

// ---------------------------------------------------------------- launch
extern "C" void kernel_launch(void* const* d_in, const int* in_sizes, int n_in,
                              void* d_out, int out_size, void* d_ws, size_t ws_size,
                              hipStream_t stream) {
  const float* means  = (const float*)d_in[0];
  const float* opac   = (const float*)d_in[1];
  const float* cols   = (const float*)d_in[2];
  const float* feats  = (const float*)d_in[3];
  const float* scales = (const float*)d_in[4];
  const float* rots   = (const float*)d_in[5];
  const float* vm     = (const float*)d_in[6];
  const float* bg     = (const float*)d_in[7];
  const unsigned char* mask = (const unsigned char*)d_in[8];
  float* out = (float*)d_out;
  float* ws  = (float*)d_ws;

  gs_preprocess<<<NG/256, 256, 0, stream>>>(means, opac, scales, rots, vm, mask,
                                            ws, out + 22*HWP);
  gs_sort_gather<<<1, 1024, 0, stream>>>(ws, feats, cols);
  gs_raster<<<HWP/256, 256, 0, stream>>>(ws, bg, out);
}
